// Encoder_2585570312714
// MI455X (gfx1250) — 8-bit matrix operand order repaired and checked by reading, compiled, not run on hardware
//
#include <hip/hip_runtime.h>

typedef __attribute__((ext_vector_type(8))) int v8i;

#define BDIM   8
#define TDIM   256
#define CDIM   16
#define DDIM   4096
#define NLEV   21
#define KREAL  336
#define KPAD   384
#define MTOT   2048

__global__ void build_A(const float* __restrict__ x, unsigned char* __restrict__ A) {
    int row = blockIdx.x * blockDim.x + threadIdx.x;
    if (row >= MTOT) return;
    unsigned char* r = A + (size_t)row * KPAD;
    unsigned int* r32 = (unsigned int*)r;
    #pragma unroll
    for (int i = 0; i < KPAD / 4; ++i) r32[i] = 0u;
    const float* xr = x + (size_t)row * CDIM;
    #pragma unroll
    for (int c = 0; c < CDIM; ++c) {
        int idx = (int)rintf(xr[c]);
        idx = idx < 0 ? 0 : (idx > NLEV - 1 ? NLEV - 1 : idx);
        r[c * NLEV + idx] = 1;
    }
}

__global__ void build_B(const float* __restrict__ lw, const float* __restrict__ cw,
                        signed char* __restrict__ Bt) {
    int d = blockIdx.x * blockDim.x + threadIdx.x;
    if (d >= DDIM) return;
    signed char* r = Bt + (size_t)d * KPAD;
    #pragma unroll
    for (int c = 0; c < CDIM; ++c) {
        float cv = cw[c * DDIM + d];
        for (int l = 0; l < NLEV; ++l) {
            float w = lw[l * DDIM + d] * cv;
            r[c * NLEV + l] = (signed char)(w > 0.0f ? 1 : (w < 0.0f ? -1 : 0));
        }
    }
    for (int k = KREAL; k < KPAD; ++k) r[k] = 0;
}

__global__ void __launch_bounds__(256) gemm_iu8(const unsigned char* __restrict__ A,
                                                const signed char*  __restrict__ Bt,
                                                signed char*        __restrict__ S) {
    const int lane = threadIdx.x & 31;
    const int wave = threadIdx.x >> 5;
    const int wg   = blockIdx.x * 8 + wave;
    const int mt   = wg >> 8;
    const int nt   = wg & 255;
    const int h    = lane >> 4;
    const int mn   = lane & 15;

    const unsigned char* arow = A  + (size_t)(mt * 16 + mn) * KPAD;
    const signed char*   brow = Bt + (size_t)(nt * 16 + mn) * KPAD;

    v8i acc = {};
    #pragma unroll
    for (int kk = 0; kk < 6; ++kk) {
        v8i a, b;
        #pragma unroll
        for (int j = 0; j < 8; ++j) {
            int ka = kk * 64 + 16 * (j >> 1) + 4 * (j & 1) + 8 * h;
            a[j] = *(const int*)(arow + ka);
            int kb = kk * 64 + 16 * (j >> 1) + 4 * (j & 1) + 8 * h;
            b[j] = *(const int*)(brow + kb);
        }
        acc = __builtin_amdgcn_wmma_i32_16x16x64_iu8(false, a, true, b, acc,
                                                     false, false);
    }

    #pragma unroll
    for (int r = 0; r < 8; ++r) {
        int m = mt * 16 + r + 8 * h;
        int n = nt * 16 + mn;
        S[(size_t)m * DDIM + n] = (signed char)acc[r];
    }
}

__global__ void ngram_sign(const signed char* __restrict__ S, float* __restrict__ out) {
    int tid = blockIdx.x * blockDim.x + threadIdx.x;
    int b = tid >> 12;
    int d = tid & (DDIM - 1);
    const signed char* sb = S + (size_t)b * TDIM * DDIM;
    int d0 = (d - 3) & (DDIM - 1);
    int d1 = (d - 2) & (DDIM - 1);
    int d2 = (d - 1) & (DDIM - 1);
    int acc = 0;
    for (int t = 0; t <= TDIM - 4; ++t) {
        int p0 = sb[(t + 0) * DDIM + d0];
        int p1 = sb[(t + 1) * DDIM + d1];
        int p2 = sb[(t + 2) * DDIM + d2];
        int p3 = sb[(t + 3) * DDIM + d];
        acc += p0 * p1 * p2 * p3;
    }
    out[tid] = acc > 0 ? 1.0f : -1.0f;
}

extern "C" void kernel_launch(void* const* d_in, const int* in_sizes, int n_in,
                              void* d_out, int out_size, void* d_ws, size_t ws_size,
                              hipStream_t stream) {
    const float* x  = (const float*)d_in[0];
    const float* lw = (const float*)d_in[1];
    const float* cw = (const float*)d_in[2];

    unsigned char* ws = (unsigned char*)d_ws;
    signed char*   S  = (signed char*)ws;
    unsigned char* A  = ws + (size_t)MTOT * DDIM;
    signed char*   Bt = (signed char*)(A + (size_t)MTOT * KPAD);

    build_A   <<<MTOT / 256, 256, 0, stream>>>(x, A);
    build_B   <<<DDIM / 256, 256, 0, stream>>>(lw, cw, Bt);
    gemm_iu8  <<<4096,       256, 0, stream>>>(A, Bt, S);
    ngram_sign<<<(BDIM * DDIM) / 256, 256, 0, stream>>>(S, (float*)d_out);
}
